// LSTMClassifier_584115552617
// MI455X (gfx1250) — matrix operand order repaired and checked by reading, not compiled, not run on hardware
//
#include <hip/hip_runtime.h>
#include <hip/hip_bf16.h>
#include <stdint.h>

// ---------------------------------------------------------------------------
// LSTM classifier for MI455X (gfx1250, wave32, WMMA + async-to-LDS).
//   B=256, T=512, I=64, H=512, G=4H=2048, O=1
// Strategy:
//   - Batch rows are independent through the recurrence: each workgroup owns
//     16 batch rows and runs the entire T=512 scan with h/c/gates in LDS,
//     workgroup barriers only (no grid sync needed).
//   - Input projection (K=64) and recurrent GEMM (K=512) fused into one
//     accumulator per 16x16 tile via v_wmma_f32_16x16x32_bf16.
//   - kt-outer / 4-accumulator-inner ordering: each A fragment loaded from
//     LDS once feeds 4 WMMAs (4x less LDS A traffic than tile-at-a-time).
//   - W_ih/W_hh pre-packed (bf16) into the wave32 B-fragment layout so each
//     lane's B operand is one contiguous 32B load; both stay L2-resident.
//   - x[:, t, :] tiles (4 KB fp32) are double-buffered into LDS with
//     global_load_async_to_lds_b128 (ASYNCcnt), prefetched one step ahead.
// ---------------------------------------------------------------------------

typedef __attribute__((ext_vector_type(16))) __bf16 v16bf;
typedef __attribute__((ext_vector_type(8)))  __bf16 v8bf;
typedef __attribute__((ext_vector_type(8)))  float  v8f;

namespace cfg {
constexpr int B = 256, T = 512, I = 64, H = 512, G = 2048;   // G = 4H
constexpr int ROWS = 16;                // batch rows per workgroup (M tile)
constexpr int NWG  = B / ROWS;          // 16 workgroups
constexpr int THREADS = 512;            // 16 wave32 waves
constexpr int NT   = G / 16;            // 128 N tiles of width 16
constexpr int KT_I = I / 32;            // 2  K tiles for W_ih
constexpr int KT_H = H / 32;            // 16 K tiles for W_hh
constexpr int NT_PER_WAVE = NT / (THREADS / 32);  // 8 (two passes of 4)
}

// ---------------------------------------------------------------------------
// Pack W (shape [G, K] fp32, used as B = W^T [K, G]) into bf16 WMMA B-matrix
// fragments. Fragment (nt, kt) covers B[kt*32 .. +31][nt*16 .. +15].
// Wave32 striping: lane l holds column n = nt*16 + (l&15); lanes 0-15 hold
// K rows kt*32+{0..7,16..23}, lanes 16-31 hold +{8..15,24..31}; elements 0-7 take the first set, 8-15 the second.
// Linear layout: out[((nt*KT + kt)*32 + lane)*16 + e]
// ---------------------------------------------------------------------------
template <int K>
__global__ void pack_wT_bf16(const float* __restrict__ W,
                             __bf16* __restrict__ out, int total) {
  int tid = blockIdx.x * blockDim.x + threadIdx.x;
  if (tid >= total) return;
  constexpr int KT = K / 32;
  int e    = tid & 15;
  int lane = (tid >> 4) & 31;
  int frag = tid >> 9;
  int kt = frag % KT;
  int nt = frag / KT;
  int n = nt * 16 + (lane & 15);
  int k = kt * 32 + ((lane < 16) ? 0 : 8) + (e & 7) + ((e < 8) ? 0 : 16);
  out[tid] = (__bf16)W[(size_t)n * K + k];
}

__device__ __forceinline__ float sigmoidf_fast(float x) {
  return 1.0f / (1.0f + __expf(-x));
}

// gfx1250 async global->LDS copy, 16 bytes per active lane (ASYNCcnt).
__device__ __forceinline__ void async_g2lds_b128(uint32_t lds_off,
                                                 const void* gptr) {
  asm volatile("global_load_async_to_lds_b128 %0, %1, off"
               :: "v"(lds_off), "v"((uint64_t)(uintptr_t)gptr)
               : "memory");
}
__device__ __forceinline__ void wait_asynccnt0() {
  asm volatile("s_wait_asynccnt 0x0" ::: "memory");
}

// A fragment (16x32 bf16) for this lane from an LDS bf16 row.
// lanes 0-15 (M=lane): K {0..7} U {16..23}; lanes 16-31: {8..15} U {24..31}.
__device__ __forceinline__ v16bf load_a_frag(const __bf16* __restrict__ row,
                                             int kbase /* kt*32 + aoff */) {
  v8bf lo = *(const v8bf*)(row + kbase);
  v8bf hi = *(const v8bf*)(row + kbase + 16);
  v16bf a;
#pragma unroll
  for (int i = 0; i < 8; ++i) { a[i] = lo[i]; a[i + 8] = hi[i]; }
  return a;
}

// Same, but source row is fp32 in LDS (x tile); convert on the fly.
__device__ __forceinline__ v16bf load_a_frag_f32(const float* __restrict__ row,
                                                 int kbase) {
  v8f lo = *(const v8f*)(row + kbase);
  v8f hi = *(const v8f*)(row + kbase + 16);
  v16bf a;
#pragma unroll
  for (int i = 0; i < 8; ++i) {
    a[i]     = (__bf16)lo[i];
    a[i + 8] = (__bf16)hi[i];
  }
  return a;
}

__global__ __launch_bounds__(cfg::THREADS)
void lstm_scan_kernel(const float* __restrict__ x,        // [B,T,I]
                      const __bf16* __restrict__ WihP,    // packed B frags
                      const __bf16* __restrict__ WhhP,    // packed B frags
                      const float* __restrict__ b_ih,     // [G]
                      const float* __restrict__ b_hh,     // [G]
                      const float* __restrict__ W_fc,     // [H]
                      const float* __restrict__ b_fc,     // [1]
                      float* __restrict__ out)            // [B]
{
  using namespace cfg;
  extern __shared__ __align__(16) char smem[];
  __bf16* h_s    = (__bf16*)smem;                    // ROWS*H bf16   (16 KB)
  float*  c_s    = (float*)(h_s + ROWS * H);         // ROWS*H f32    (32 KB)
  float*  g_s    = c_s + ROWS * H;                   // ROWS*G f32   (128 KB)
  float*  bias_s = g_s + ROWS * G;                   // G f32         ( 8 KB)
  float*  x_s    = bias_s + G;                       // 2*ROWS*I f32  ( 8 KB)

  const int tid  = threadIdx.x;
  const int lane = tid & 31;
  const int wave = tid >> 5;
  const int b0   = blockIdx.x * ROWS;

  // --- init h = 0, c = 0, bias = b_ih + b_hh ---
  for (int idx = tid; idx < ROWS * H; idx += THREADS) {
    h_s[idx] = (__bf16)0.0f;
    c_s[idx] = 0.0f;
  }
  for (int j = tid; j < G; j += THREADS) bias_s[j] = b_ih[j] + b_hh[j];

  // --- async prefetch x tile for t=0 into buffer 0 ---
  // Tile = ROWS*I fp32 = 4 KB = 256 x 16B chunks; threads 0..255 (waves 0-7).
  const uint32_t x_lds_base = (uint32_t)(uintptr_t)x_s;   // LDS byte offset
  const int chunk_r = tid >> 4;          // row 0..15
  const int chunk_c = (tid & 15) * 4;    // fp32 offset within row (16B chunk)
  const float* xrow = x + (size_t)(b0 + chunk_r) * (T * I) + chunk_c;
  const uint32_t lds_chunk = x_lds_base + (uint32_t)(chunk_r * I + chunk_c) * 4u;
  if (wave < 8) {
    async_g2lds_b128(lds_chunk, xrow);   // t = 0, buffer 0
  }

  const int nlo   = lane & 15;
  const int aoff  = (lane < 16) ? 0 : 8;   // A-fragment K sub-offset
  const int mbase = (lane < 16) ? 0 : 8;   // C/D-fragment M base
  const __bf16* arow_h = h_s + nlo * H;    // A row for this lane (h tile)

  __syncthreads();   // bias/h/c visible (async tile waited below)

  for (int t = 0; t < T; ++t) {
    const int buf = t & 1;
    // --- make x tile t visible to all waves ---
    if (wave < 8) wait_asynccnt0();
    __syncthreads();
    // --- prefetch tile t+1 into the other buffer (overlaps GEMM) ---
    if (wave < 8 && (t + 1) < T) {
      async_g2lds_b128(x_lds_base + (uint32_t)((buf ^ 1) * ROWS * I + chunk_r * I + chunk_c) * 4u,
                       xrow + (size_t)(t + 1) * I);
    }
    const float* arow_x = x_s + buf * ROWS * I + nlo * I;

    // --- gates = x_t @ W_ih^T + h @ W_hh^T ---
    // Two passes of 4 n-tiles: each A fragment feeds 4 WMMAs.
#pragma unroll
    for (int half = 0; half < 2; ++half) {
      const int nt0 = wave * NT_PER_WAVE + half * 4;
      v8f acc0 = {}, acc1 = {}, acc2 = {}, acc3 = {};

      // input projection, K = 64 (fp32 in LDS, cvt on the fly)
#pragma unroll
      for (int kt = 0; kt < KT_I; ++kt) {
        v16bf a = load_a_frag_f32(arow_x, kt * 32 + aoff);
        const __bf16* bp = WihP + ((size_t)(nt0 * KT_I + kt) * 32 + lane) * 16;
        v16bf bb0 = *(const v16bf*)(bp);
        v16bf bb1 = *(const v16bf*)(bp + (size_t)KT_I * 512);
        v16bf bb2 = *(const v16bf*)(bp + (size_t)2 * KT_I * 512);
        v16bf bb3 = *(const v16bf*)(bp + (size_t)3 * KT_I * 512);
        acc0 = __builtin_amdgcn_wmma_f32_16x16x32_bf16(false, a, false, bb0, (short)0, acc0, false, false);
        acc1 = __builtin_amdgcn_wmma_f32_16x16x32_bf16(false, a, false, bb1, (short)0, acc1, false, false);
        acc2 = __builtin_amdgcn_wmma_f32_16x16x32_bf16(false, a, false, bb2, (short)0, acc2, false, false);
        acc3 = __builtin_amdgcn_wmma_f32_16x16x32_bf16(false, a, false, bb3, (short)0, acc3, false, false);
      }
      // recurrent projection, K = 512
#pragma unroll 4
      for (int kt = 0; kt < KT_H; ++kt) {
        v16bf a = load_a_frag(arow_h, kt * 32 + aoff);
        const __bf16* bp = WhhP + ((size_t)(nt0 * KT_H + kt) * 32 + lane) * 16;
        v16bf bb0 = *(const v16bf*)(bp);
        v16bf bb1 = *(const v16bf*)(bp + (size_t)KT_H * 512);
        v16bf bb2 = *(const v16bf*)(bp + (size_t)2 * KT_H * 512);
        v16bf bb3 = *(const v16bf*)(bp + (size_t)3 * KT_H * 512);
        acc0 = __builtin_amdgcn_wmma_f32_16x16x32_bf16(false, a, false, bb0, (short)0, acc0, false, false);
        acc1 = __builtin_amdgcn_wmma_f32_16x16x32_bf16(false, a, false, bb1, (short)0, acc1, false, false);
        acc2 = __builtin_amdgcn_wmma_f32_16x16x32_bf16(false, a, false, bb2, (short)0, acc2, false, false);
        acc3 = __builtin_amdgcn_wmma_f32_16x16x32_bf16(false, a, false, bb3, (short)0, acc3, false, false);
      }

      // D layout: VGPR r -> M = mbase + r, N = nt*16 + (lane&15)
#pragma unroll
      for (int r = 0; r < 8; ++r) {
        const size_t rowoff = (size_t)(mbase + r) * G + nlo;
        g_s[rowoff + (nt0 + 0) * 16] = acc0[r];
        g_s[rowoff + (nt0 + 1) * 16] = acc1[r];
        g_s[rowoff + (nt0 + 2) * 16] = acc2[r];
        g_s[rowoff + (nt0 + 3) * 16] = acc3[r];
      }
    }
    __syncthreads();

    // --- elementwise cell update (PyTorch gate order i,f,g,o) ---
    for (int idx = tid; idx < ROWS * H; idx += THREADS) {
      int r = idx >> 9;          // / H
      int j = idx & (H - 1);     // % H
      const float* gr = g_s + (size_t)r * G;
      float ig = sigmoidf_fast(gr[j]         + bias_s[j]);
      float fg = sigmoidf_fast(gr[H + j]     + bias_s[H + j]);
      float gg = tanhf(        gr[2 * H + j] + bias_s[2 * H + j]);
      float og = sigmoidf_fast(gr[3 * H + j] + bias_s[3 * H + j]);
      float c = fg * c_s[idx] + ig * gg;
      c_s[idx] = c;
      h_s[idx] = (__bf16)(og * tanhf(c));
    }
    __syncthreads();
  }

  // --- final FC: out[b] = h[b,:] . W_fc + b_fc ; wave w handles row w ---
  {
    const int r = wave;
    float partial = 0.0f;
    for (int j = lane; j < H; j += 32)
      partial += (float)h_s[(size_t)r * H + j] * W_fc[j];
#pragma unroll
    for (int off = 16; off > 0; off >>= 1)
      partial += __shfl_down(partial, off, 32);
    if (lane == 0) out[b0 + r] = partial + b_fc[0];
  }
}

// ---------------------------------------------------------------------------
extern "C" void kernel_launch(void* const* d_in, const int* in_sizes, int n_in,
                              void* d_out, int out_size, void* d_ws, size_t ws_size,
                              hipStream_t stream) {
  using namespace cfg;
  const float* x    = (const float*)d_in[0];
  const float* W_ih = (const float*)d_in[1];
  const float* W_hh = (const float*)d_in[2];
  const float* b_ih = (const float*)d_in[3];
  const float* b_hh = (const float*)d_in[4];
  const float* W_fc = (const float*)d_in[5];
  const float* b_fc = (const float*)d_in[6];
  float* out = (float*)d_out;

  // Workspace: packed bf16 weight fragments (W_ih: 256 KB, W_hh: 2 MB).
  __bf16* WihP = (__bf16*)d_ws;
  __bf16* WhhP = WihP + (size_t)NT * KT_I * 512;

  const int totIh = NT * KT_I * 512;   // 131072
  const int totHh = NT * KT_H * 512;   // 1048576
  pack_wT_bf16<I><<<totIh / 256, 256, 0, stream>>>(W_ih, WihP, totIh);
  pack_wT_bf16<H><<<totHh / 256, 256, 0, stream>>>(W_hh, WhhP, totHh);

  const size_t smem_bytes =
      (size_t)ROWS * H * sizeof(__bf16) +       // h (bf16)
      (size_t)ROWS * H * sizeof(float)  +       // c
      (size_t)ROWS * G * sizeof(float)  +       // gates
      (size_t)G * sizeof(float)         +       // bias
      (size_t)2 * ROWS * I * sizeof(float);     // x double buffer (~192 KB)

  lstm_scan_kernel<<<NWG, THREADS, smem_bytes, stream>>>(
      x, WihP, WhhP, b_ih, b_hh, W_fc, b_fc, out);
}
